// PolymerGNN_Joint_23639499997301
// MI455X (gfx1250) — matrix operand order repaired and checked by reading, not compiled, not run on hardware
//
#include <hip/hip_runtime.h>
#include <hip/hip_bf16.h>
#include <math.h>

// ---------------------------------------------------------------------------
// PolymerGNN joint forward for gfx1250 (MI455X).
// GEMMs use v_wmma_f32_16x16x32_bf16 (wave32, f32 accumulate).
// Scatter/segment ops use L2-resident float atomics (bit-trick max).
// Top-k (k=N/2) via 4-pass radix select on float key bits.
// Workspace use: 3*N*H + 5*N floats + small stats/state  ~= 78 MB.
// ---------------------------------------------------------------------------

#define NNODE 50000
#define NEDGE 600000
#define ETOT  (NEDGE + NNODE)   // edges + self loops for GAT
#define FIN   64
#define HDIM  128
#define KSEL  ((NNODE + 1) / 2) // 25000
#define NH    ((size_t)NNODE * HDIM)

typedef __attribute__((ext_vector_type(16))) __bf16 v16bf;
typedef __attribute__((ext_vector_type(8)))  float  v8f;

// -------------------------------- helpers ----------------------------------

__device__ inline float fixfin(float v) {
  // where(isfinite(v), v, 0)
  return (fabsf(v) < __builtin_inff()) ? v : 0.0f;
}

__device__ inline void atomicMaxFloat(float* addr, float val) {
  val = val + 0.0f;              // canonicalize -0 -> +0
  if (val >= 0.0f)
    atomicMax((int*)addr, __float_as_int(val));
  else
    atomicMin((unsigned int*)addr, __float_as_uint(val));
}

__device__ inline unsigned fkey(float f) {
  unsigned u = __float_as_uint(f);
  return (u & 0x80000000u) ? ~u : (u | 0x80000000u);
}

// ----------------------------- WMMA tile I/O -------------------------------
// 16-bit A-matrix 16x32 (MxK): lanes 0-15 hold M=lane; VGPR0-3 = K 0..7 (lo
// half) / 8..15 (hi half lanes 16-31), VGPR4-7 = K 16..23 / 24..31, 2 packed
// bf16 per VGPR.
__device__ inline v16bf load_a_tile(const float* A, int ld, int row0, int k0) {
  int lane = threadIdx.x & 31;
  int m    = lane & 15;
  int half = lane >> 4;
  const float* base = A + (size_t)(row0 + m) * ld + k0;
  v16bf a;
#pragma unroll
  for (int e = 0; e < 16; ++e) {
    int v = e >> 1, p = e & 1;
    int k = (v < 4) ? (half * 8 + v * 2 + p) : (16 + half * 8 + (v - 4) * 2 + p);
    a[e] = (__bf16)base[k];
  }
  return a;
}

// 16-bit B-matrix 32x16 (KxN): lane covers column N=lane%16; elements 0-7 hold
// K=8*half+0..7, elements 8-15 hold K=16+8*half+0..7 (same K order as A).
__device__ inline v16bf load_b_tile(const float* B, int ld, int k0, int col0) {
  int lane = threadIdx.x & 31;
  int n    = lane & 15;
  int half = lane >> 4;
  const float* base = B + (size_t)k0 * ld + col0 + n;
  v16bf b;
#pragma unroll
  for (int e = 0; e < 16; ++e) b[e] = (__bf16)base[(size_t)((e < 8) ? (half * 8 + e) : (16 + half * 8 + e - 8)) * ld];
  return b;
}

// 32-bit C/D 16x16: lane l, VGPR v -> row = v + 8*(l/16), col = l%16
__device__ inline void store_c_tile(float* C, int ld, int row0, int col0,
                                    v8f acc, const float* bias) {
  int lane  = threadIdx.x & 31;
  int n     = lane & 15;
  int mbase = (lane >> 4) * 8;
  float badd = bias ? bias[col0 + n] : 0.0f;
#pragma unroll
  for (int v = 0; v < 8; ++v)
    C[(size_t)(row0 + mbase + v) * ld + col0 + n] = acc[v] + badd;
}

// ------------------------------ GEMM kernels -------------------------------

// h = x @ gat_W   (x:[N,64] f32, W:[64,128] f32, h:[N,128] f32)
__global__ void __launch_bounds__(32)
gemm_gat_k(const float* __restrict__ X, const float* __restrict__ W,
           float* __restrict__ Hout) {
  int row0 = blockIdx.x * 16, col0 = blockIdx.y * 16;
  v8f acc = {};
#pragma unroll
  for (int kt = 0; kt < 2; ++kt) {
    v16bf a = load_a_tile(X, FIN, row0, kt * 32);
    v16bf b = load_b_tile(W, HDIM, kt * 32, col0);
    acc = __builtin_amdgcn_wmma_f32_16x16x32_bf16(false, a, false, b,
                                                  (short)0, acc, false, false);
  }
  store_c_tile(Hout, HDIM, row0, col0, acc, nullptr);
}

// z = agg @ Wl + bl + y @ Wr   (all [N,128] x [128,128])
__global__ void __launch_bounds__(32)
gemm_sage_k(const float* __restrict__ AGG, const float* __restrict__ Y,
            const float* __restrict__ Wl, const float* __restrict__ Wr,
            const float* __restrict__ bl, float* __restrict__ Z) {
  int row0 = blockIdx.x * 16, col0 = blockIdx.y * 16;
  v8f acc = {};
#pragma unroll
  for (int kt = 0; kt < 4; ++kt) {
    v16bf a = load_a_tile(AGG, HDIM, row0, kt * 32);
    v16bf b = load_b_tile(Wl, HDIM, kt * 32, col0);
    acc = __builtin_amdgcn_wmma_f32_16x16x32_bf16(false, a, false, b,
                                                  (short)0, acc, false, false);
  }
#pragma unroll
  for (int kt = 0; kt < 4; ++kt) {
    v16bf a = load_a_tile(Y, HDIM, row0, kt * 32);
    v16bf b = load_b_tile(Wr, HDIM, kt * 32, col0);
    acc = __builtin_amdgcn_wmma_f32_16x16x32_bf16(false, a, false, b,
                                                  (short)0, acc, false, false);
  }
  store_c_tile(Z, HDIM, row0, col0, acc, bl);
}

// ---------------------------- elementwise kernels --------------------------

__global__ void fill_f32_k(float* p, size_t n, float v) {
  size_t i = (size_t)blockIdx.x * blockDim.x + threadIdx.x;
  size_t stride = (size_t)gridDim.x * blockDim.x;
  for (; i < n; i += stride) p[i] = v;
}

// out = where(isfinite(out), out, 0) + bias[c]
__global__ void fix_bias_k(float* x, const float* __restrict__ b, size_t n) {
  size_t i = (size_t)blockIdx.x * blockDim.x + threadIdx.x;
  if (i >= n) return;
  x[i] = fixfin(x[i]) + b[i & (HDIM - 1)];
}

__global__ void fix_zero_k(float* x, size_t n) {
  size_t i = (size_t)blockIdx.x * blockDim.x + threadIdx.x;
  if (i >= n) return;
  x[i] = fixfin(x[i]);
}

// ------------------------------ GAT attention ------------------------------

// per-node: s_src[n] = h[n].att_src ; s_dst[n] = h[n].att_dst
__global__ void att_scores_k(const float* __restrict__ h,
                             const float* __restrict__ asrc,
                             const float* __restrict__ adst,
                             float* __restrict__ ssrc, float* __restrict__ sdst) {
  int i = blockIdx.x * blockDim.x + threadIdx.x;
  if (i >= NNODE) return;
  const float* hp = h + (size_t)i * HDIM;
  float a = 0.f, b = 0.f;
#pragma unroll 4
  for (int c = 0; c < HDIM; ++c) { float v = hp[c]; a += v * asrc[c]; b += v * adst[c]; }
  ssrc[i] = a; sdst[i] = b;
}

__device__ inline void edge_sd(int i, const int* src, const int* dst, int& s, int& d) {
  if (i < NEDGE) { s = src[i]; d = dst[i]; }
  else           { s = d = i - NEDGE; }     // self loops appended
}

__global__ void gat_edge_max_k(const int* __restrict__ src, const int* __restrict__ dst,
                               const float* __restrict__ ssrc, const float* __restrict__ sdst,
                               float* __restrict__ emax) {
  int i = blockIdx.x * blockDim.x + threadIdx.x;
  if (i >= ETOT) return;
  int s, d; edge_sd(i, src, dst, s, d);
  float e = ssrc[s] + sdst[d];
  e = (e > 0.f) ? e : 0.2f * e;            // leaky_relu(0.2)
  atomicMaxFloat(&emax[d], e);
}

__global__ void gat_edge_sum_k(const int* __restrict__ src, const int* __restrict__ dst,
                               const float* __restrict__ ssrc, const float* __restrict__ sdst,
                               const float* __restrict__ emax, float* __restrict__ denom) {
  int i = blockIdx.x * blockDim.x + threadIdx.x;
  if (i >= ETOT) return;
  int s, d; edge_sd(i, src, dst, s, d);
  float e = ssrc[s] + sdst[d];
  e = (e > 0.f) ? e : 0.2f * e;
  float m = fixfin(emax[d]);
  atomicAdd(&denom[d], expf(e - m));
}

// out[d][c] = max over edges of h[s][c] * alpha   (block = 1 edge x 128 ch)
__global__ void gat_edge_msg_k(const int* __restrict__ src, const int* __restrict__ dst,
                               const float* __restrict__ ssrc, const float* __restrict__ sdst,
                               const float* __restrict__ emax, const float* __restrict__ denom,
                               const float* __restrict__ h, float* __restrict__ out) {
  __shared__ int sh_s, sh_d;
  __shared__ float sh_alpha;
  int i = blockIdx.x;
  if (threadIdx.x == 0) {
    int s, d; edge_sd(i, src, dst, s, d);
    float e = ssrc[s] + sdst[d];
    e = (e > 0.f) ? e : 0.2f * e;
    float m  = fixfin(emax[d]);
    float ex = expf(e - m);
    sh_alpha = ex / (denom[d] + 1e-16f);
    sh_s = s; sh_d = d;
  }
  __syncthreads();
  int c = threadIdx.x;
  atomicMaxFloat(&out[(size_t)sh_d * HDIM + c], h[(size_t)sh_s * HDIM + c] * sh_alpha);
}

// ------------------------------ BatchNorm ----------------------------------

__global__ void bn_stats_k(const float* __restrict__ x, float* __restrict__ sum,
                           float* __restrict__ sq) {
  int c = threadIdx.x;                          // 128 threads
  int rows_per = (NNODE + gridDim.x - 1) / gridDim.x;
  int r0 = blockIdx.x * rows_per;
  int r1 = r0 + rows_per; if (r1 > NNODE) r1 = NNODE;
  float s = 0.f, q = 0.f;
  for (int r = r0; r < r1; ++r) { float v = x[(size_t)r * HDIM + c]; s += v; q += v * v; }
  atomicAdd(&sum[c], s); atomicAdd(&sq[c], q);
}

__global__ void bn_finalize_k(const float* __restrict__ sum, const float* __restrict__ sq,
                              const float* __restrict__ g, const float* __restrict__ b,
                              float* __restrict__ scale, float* __restrict__ shift) {
  int c = threadIdx.x;
  float mu  = sum[c] / (float)NNODE;
  float var = sq[c] / (float)NNODE - mu * mu;
  float s   = g[c] * rsqrtf(var + 1e-5f);
  scale[c] = s; shift[c] = b[c] - mu * s;
}

__global__ void bn_prelu_k(float* x, const float* __restrict__ scale,
                           const float* __restrict__ shift, const float* __restrict__ ap,
                           size_t n) {
  size_t i = (size_t)blockIdx.x * blockDim.x + threadIdx.x;
  if (i >= n) return;
  int c = (int)(i & (HDIM - 1));
  float v = x[i] * scale[c] + shift[c];
  float a = ap[0];
  x[i] = (v >= 0.f) ? v : a * v;
}

// -------------------------- SAGE / pool scatters ---------------------------

__global__ void sage_edge_max_k(const int* __restrict__ src, const int* __restrict__ dst,
                                const float* __restrict__ y, float* __restrict__ agg) {
  __shared__ int sh_s, sh_d;
  if (threadIdx.x == 0) { sh_s = src[blockIdx.x]; sh_d = dst[blockIdx.x]; }
  __syncthreads();
  int c = threadIdx.x;
  atomicMaxFloat(&agg[(size_t)sh_d * HDIM + c], y[(size_t)sh_s * HDIM + c]);
}

__global__ void pool_edge_sum_k(const int* __restrict__ src, const int* __restrict__ dst,
                                const float* __restrict__ y, float* __restrict__ agg) {
  __shared__ int sh_s, sh_d;
  if (threadIdx.x == 0) { sh_s = src[blockIdx.x]; sh_d = dst[blockIdx.x]; }
  __syncthreads();
  int c = threadIdx.x;
  atomicAdd(&agg[(size_t)sh_d * HDIM + c], y[(size_t)sh_s * HDIM + c]);
}

__global__ void pool_score_k(const float* __restrict__ agg, const float* __restrict__ y,
                             const float* __restrict__ wrel, const float* __restrict__ brel,
                             const float* __restrict__ wroot, float* __restrict__ score) {
  int i = blockIdx.x * blockDim.x + threadIdx.x;
  if (i >= NNODE) return;
  const float* ap = agg + (size_t)i * HDIM;
  const float* yp = y   + (size_t)i * HDIM;
  float a = 0.f, b = 0.f;
#pragma unroll 4
  for (int c = 0; c < HDIM; ++c) { a += ap[c] * wrel[c]; b += yp[c] * wroot[c]; }
  score[i] = a + brel[0] + b;
}

// ------------------------- radix top-k (k = 25000) -------------------------
// sel[0]=prefix(u32) sel[1]=kneed sel[2]=tie counter, sel[4..259]=histogram

__global__ void sel_init_k(int* sel) {
  int t = threadIdx.x;
  sel[4 + t] = 0;
  if (t == 0) { sel[0] = 0; sel[1] = KSEL; sel[2] = 0; }
}

__global__ void sel_hist_k(const float* __restrict__ score, int* sel, int pass) {
  __shared__ int lh[256];
  int t = threadIdx.x;
  lh[t] = 0;
  __syncthreads();
  unsigned prefix = (unsigned)sel[0];
  int shift = 24 - 8 * pass;
  unsigned maskHi = pass ? (0xFFFFFFFFu << (32 - 8 * pass)) : 0u;
  for (int i = blockIdx.x * 256 + t; i < NNODE; i += gridDim.x * 256) {
    unsigned u = fkey(score[i]);
    if ((u & maskHi) == prefix) atomicAdd(&lh[(u >> shift) & 255], 1);
  }
  __syncthreads();
  if (lh[t]) atomicAdd(&sel[4 + t], lh[t]);
}

__global__ void sel_scan_k(int* sel, int pass) {
  int shift = 24 - 8 * pass;
  int kneed = sel[1];
  int cum = 0, b;
  for (b = 255; b > 0; --b) {
    int h = sel[4 + b];
    if (cum + h >= kneed) break;
    cum += h;
  }
  sel[0] = (int)((unsigned)sel[0] | ((unsigned)b << shift));
  sel[1] = kneed - cum;
  for (int i = 0; i < 256; ++i) sel[4 + i] = 0;
}

// select nodes with key > T (ties up to kneed), embed[c] = max(y2[n][c]*tanh(score[n]))
__global__ void sel_apply_k(const float* __restrict__ score, const float* __restrict__ y2,
                            int* sel, float* __restrict__ embed) {
  __shared__ float sh_t;
  __shared__ int sh_inc;
  int n = blockIdx.x;
  if (threadIdx.x == 0) {
    float sc = score[n];
    unsigned u = fkey(sc);
    unsigned T = (unsigned)sel[0];
    int inc = 0;
    if (u > T) inc = 1;
    else if (u == T && atomicAdd(&sel[2], 1) < sel[1]) inc = 1;
    sh_inc = inc;
    sh_t = tanhf(sc);
  }
  __syncthreads();
  if (sh_inc) {
    int c = threadIdx.x;
    atomicMaxFloat(&embed[c], y2[(size_t)n * HDIM + c] * sh_t);
  }
}

// --------------------------------- head ------------------------------------

__global__ void head_k(const float* eA, const float* eG, const float* add,
                       const float* fc1W, const float* fc1b, const float* a1p,
                       const float* fc2W1, const float* fc2b1, const float* a2p,
                       const float* fc2W2, const float* fc2b2,
                       const float* fc3W, const float* fc3b,
                       const float* multW, const float* multb,
                       float* out) {
  __shared__ float z[2 * HDIM + 8], x1[HDIM], uu[HDIM];
  __shared__ float r0[HDIM], r1[HDIM], r2[HDIM];
  int t = threadIdx.x;
  z[t] = eA[t]; z[HDIM + t] = eG[t];
  if (t < 8) z[2 * HDIM + t] = add[t];
  __syncthreads();
  float a1 = a1p[0];
  float acc = fc1b[t];
  for (int i = 0; i < 2 * HDIM + 8; ++i) acc += z[i] * fc1W[i * HDIM + t];
  x1[t] = (acc >= 0.f) ? acc : a1 * acc;
  __syncthreads();
  float a2 = a2p[0];
  float acc2 = fc2b1[t];
  for (int i = 0; i < HDIM; ++i) acc2 += x1[i] * fc2W1[i * HDIM + t];
  uu[t] = (acc2 >= 0.f) ? acc2 : a2 * acc2;
  __syncthreads();
  r0[t] = uu[t] * fc2W2[t];
  r1[t] = x1[t] * fc3W[t];
  r2[t] = x1[t] * multW[t];
  __syncthreads();
  for (int s = HDIM / 2; s > 0; s >>= 1) {
    if (t < s) { r0[t] += r0[t + s]; r1[t] += r1[t + s]; r2[t] += r2[t + s]; }
    __syncthreads();
  }
  if (t == 0) {
    out[0] = r0[0] + fc2b2[0];                                  // x_IV
    out[1] = expf(r1[0] + fc3b[0]) * tanhf(r2[0] + multb[0]);   // x_Tg
  }
}

// --------------------------------- host ------------------------------------

static inline void fillf(float* p, size_t n, float v, hipStream_t st) {
  unsigned g = (unsigned)((n + 255) / 256);
  if (g > (1u << 16)) g = 1u << 16;
  fill_f32_k<<<dim3(g), dim3(256), 0, st>>>(p, n, v);
}

static void run_branch(const float* x, const int* eidx, const float* const* P,
                       float* embed, float* B0, float* B1, float* B2,
                       float* ssrc, float* sdst, float* emax, float* denom,
                       float* score, float* bnsum, float* bnsq,
                       float* bnscale, float* bnshift, int* sel, hipStream_t st) {
  const float NEG_INF = -__builtin_inff();
  const int* esrc = eidx;
  const int* edst = eidx + NEDGE;
  const unsigned nhb = (unsigned)((NH + 255) / 256);

  // GAT: h = x @ gat_W  (WMMA)
  gemm_gat_k<<<dim3(NNODE / 16, HDIM / 16), 32, 0, st>>>(x, P[0], B0);
  att_scores_k<<<(NNODE + 255) / 256, 256, 0, st>>>(B0, P[1], P[2], ssrc, sdst);
  fillf(emax, NNODE, NEG_INF, st);
  fillf(denom, NNODE, 0.f, st);
  fillf(B1, NH, NEG_INF, st);
  gat_edge_max_k<<<(ETOT + 255) / 256, 256, 0, st>>>(esrc, edst, ssrc, sdst, emax);
  gat_edge_sum_k<<<(ETOT + 255) / 256, 256, 0, st>>>(esrc, edst, ssrc, sdst, emax, denom);
  gat_edge_msg_k<<<ETOT, HDIM, 0, st>>>(esrc, edst, ssrc, sdst, emax, denom, B0, B1);
  fix_bias_k<<<nhb, 256, 0, st>>>(B1, P[3], NH);

  // BN1 + PReLU
  fillf(bnsum, HDIM, 0.f, st); fillf(bnsq, HDIM, 0.f, st);
  bn_stats_k<<<256, HDIM, 0, st>>>(B1, bnsum, bnsq);
  bn_finalize_k<<<1, HDIM, 0, st>>>(bnsum, bnsq, P[4], P[5], bnscale, bnshift);
  bn_prelu_k<<<nhb, 256, 0, st>>>(B1, bnscale, bnshift, P[6], NH);

  // SAGE: agg = segment_max(y[src]); z = agg@Wl + bl + y@Wr  (WMMA)
  fillf(B2, NH, NEG_INF, st);
  sage_edge_max_k<<<NEDGE, HDIM, 0, st>>>(esrc, edst, B1, B2);
  fix_zero_k<<<nhb, 256, 0, st>>>(B2, NH);
  gemm_sage_k<<<dim3(NNODE / 16, HDIM / 16), 32, 0, st>>>(B2, B1, P[7], P[9], P[8], B0);

  // BN2 + PReLU
  fillf(bnsum, HDIM, 0.f, st); fillf(bnsq, HDIM, 0.f, st);
  bn_stats_k<<<256, HDIM, 0, st>>>(B0, bnsum, bnsq);
  bn_finalize_k<<<1, HDIM, 0, st>>>(bnsum, bnsq, P[10], P[11], bnscale, bnshift);
  bn_prelu_k<<<nhb, 256, 0, st>>>(B0, bnscale, bnshift, P[12], NH);

  // SAGPool: score, top-k select, embed = max over selected rows
  fillf(B2, NH, 0.f, st);
  pool_edge_sum_k<<<NEDGE, HDIM, 0, st>>>(esrc, edst, B0, B2);
  pool_score_k<<<(NNODE + 255) / 256, 256, 0, st>>>(B2, B0, P[13], P[14], P[15], score);
  sel_init_k<<<1, 256, 0, st>>>(sel);
  for (int p = 0; p < 4; ++p) {
    sel_hist_k<<<256, 256, 0, st>>>(score, sel, p);
    sel_scan_k<<<1, 1, 0, st>>>(sel, p);
  }
  fillf(embed, HDIM, NEG_INF, st);
  sel_apply_k<<<NNODE, HDIM, 0, st>>>(score, B0, sel, embed);
}

extern "C" void kernel_launch(void* const* d_in, const int* in_sizes, int n_in,
                              void* d_out, int out_size, void* d_ws, size_t ws_size,
                              hipStream_t stream) {
  (void)in_sizes; (void)n_in; (void)out_size; (void)ws_size;
  // Flat input order: top-level dict insertion order; nested param dicts in
  // insertion order (16 leaves per branch, 12 head leaves).
  const float* Ax  = (const float*)d_in[0];
  const int*   Ae  = (const int*)d_in[1];
  const float* Gx  = (const float*)d_in[2];
  const int*   Ge  = (const int*)d_in[3];
  const float* add = (const float*)d_in[4];
  const float* AP[16]; for (int i = 0; i < 16; ++i) AP[i] = (const float*)d_in[5 + i];
  const float* GP[16]; for (int i = 0; i < 16; ++i) GP[i] = (const float*)d_in[21 + i];
  const float* HP[12]; for (int i = 0; i < 12; ++i) HP[i] = (const float*)d_in[37 + i];

  float* ws = (float*)d_ws;
  float* B0     = ws;              // h -> z -> y2
  float* B1     = B0 + NH;         // gat out -> y (post BN1/PReLU)
  float* B2     = B1 + NH;         // agg (max) -> aggsum
  float* ssrc   = B2 + NH;
  float* sdst   = ssrc + NNODE;
  float* emax   = sdst + NNODE;
  float* denom  = emax + NNODE;
  float* score  = denom + NNODE;
  float* embedA = score + NNODE;
  float* embedG = embedA + HDIM;
  float* bnsum  = embedG + HDIM;
  float* bnsq   = bnsum + HDIM;
  float* bnscale = bnsq + HDIM;
  float* bnshift = bnscale + HDIM;
  int*   sel    = (int*)(bnshift + HDIM);   // 260 ints

  run_branch(Ax, Ae, AP, embedA, B0, B1, B2, ssrc, sdst, emax, denom,
             score, bnsum, bnsq, bnscale, bnshift, sel, stream);
  run_branch(Gx, Ge, GP, embedG, B0, B1, B2, ssrc, sdst, emax, denom,
             score, bnsum, bnsq, bnscale, bnshift, sel, stream);

  head_k<<<1, HDIM, 0, stream>>>(embedA, embedG, add,
                                 HP[0], HP[1], HP[2], HP[3], HP[4], HP[5],
                                 HP[6], HP[7], HP[8], HP[9], HP[10], HP[11],
                                 (float*)d_out);
}
